// LSTMClassifier_33148557591023
// MI455X (gfx1250) — compile-verified, matrix operand order checked by reading, not run on hardware
//
#include <hip/hip_runtime.h>
#include <hip/hip_bf16.h>

typedef __attribute__((ext_vector_type(16))) __bf16 v16bf;
typedef __attribute__((ext_vector_type(8)))  float  v8f;

#define T_STEPS 2048
#define HID     256
#define GATES   1024
#define INP     6
#define BT      16      // batch rows per workgroup
#define HSTR    264     // padded bf16 stride for h in LDS (bank spread)
#define NTHREADS 512    // 16 waves: 4 waves/SIMD for latency hiding

// Packed weight buffer in d_ws (bf16 units):
//   [0, 262144)        W_hh B-fragments: 64 ntg x 8 ks x (32 lanes x 16)
//   [262144, +32768)   W_ih(+bias) B-fragments, K padded 6->32:
//                      rows 0..5 = W_ih, row 6 = b_ih+b_hh, rows 7..31 = 0
#define WHH_ELEMS  262144
#define WIH_ELEMS  32768
#define PK_ELEMS   (WHH_ELEMS + WIH_ELEMS)

// Dynamic LDS layout (bytes):
//   [0, 262144)        bf16 B-cache: ks<4 half of W_hh fragments
//   [262144, +8448)    h  bf16 [16][264]
//   [270592, +384)     x_t f32 [16][6]
#define BC_BYTES   262144
#define H_OFF      262144
#define XS_OFF     270592
#define SMEM_BYTES 270976

// ---------------------------------------------------------------------------
// Pack W_hh and W_ih(+bias) into bf16 WMMA B-fragment order.
// B tile is 32x16 (KxN): lanes 0-15: N=lane, K pairs
// {0,1},{2,3},{4,5},{6,7},{16,17},{18,19},{20,21},{22,23}; lanes 16-31 same +8.
// ---------------------------------------------------------------------------
__global__ __launch_bounds__(256) void pack_weights_kernel(
    const float* __restrict__ W_hh, const float* __restrict__ W_ih,
    const float* __restrict__ b_ih, const float* __restrict__ b_hh,
    __bf16* __restrict__ pk)
{
    int idx  = blockIdx.x * 256 + threadIdx.x;     // PK_ELEMS total
    if (idx >= PK_ELEMS) return;
    int e    = idx & 15;
    int lane = (idx >> 4) & 31;
    int p    = e >> 1;
    int lo   = e & 1;
    int hi   = (lane >= 16) ? 1 : 0;
    int nl   = lane & 15;
    int k_off = ((p < 4) ? (2 * p) : (16 + 2 * (p - 4))) + 8 * hi + lo;
    if (idx < WHH_ELEMS) {
        int frag = idx >> 9;
        int ks   = frag & 7;
        int nt   = frag >> 3;
        int n = nt * 16 + nl;
        int k = ks * 32 + k_off;
        pk[idx] = (__bf16)W_hh[n * HID + k];       // B[k][n] = W_hh[n][k]
    } else {
        int ntg = (idx - WHH_ELEMS) >> 9;          // 0..63
        int n   = ntg * 16 + nl;
        float v = 0.0f;
        if (k_off < INP)       v = W_ih[n * INP + k_off];
        else if (k_off == INP) v = b_ih[n] + b_hh[n];   // bias row (A has 1.0)
        pk[idx] = (__bf16)v;
    }
}

// ---------------------------------------------------------------------------
// Fused LSTM recurrence + heads. 8 WGs x 512 threads (16 waves).
// Wave w owns tiles ntg = g*16 + w for g in {0..3}  ->  all four gates of
// hidden units [w*16, w*16+16)  ->  register-local c/h update (8 c / lane).
// Per tile per step: 1 WMMA for xg+bias (K-padded x_t) + 8 WMMAs for h@W_hh^T.
// ---------------------------------------------------------------------------
__global__ __launch_bounds__(NTHREADS) void lstm_fused_kernel(
    const float* __restrict__ x,
    const __bf16* __restrict__ Wpk,
    const float* __restrict__ Wg, const float* __restrict__ bg,
    const float* __restrict__ Wh, const float* __restrict__ bh,
    const float* __restrict__ Wy, const float* __restrict__ by,
    const float* __restrict__ Wl, const float* __restrict__ bl,
    float* __restrict__ out)
{
    extern __shared__ char smem[];
    __bf16* bc    = (__bf16*)smem;
    __bf16* h_lds = (__bf16*)(smem + H_OFF);
    float*  x_s   = (float*)(smem + XS_OFF);

    const int tid  = threadIdx.x;
    const int lane = tid & 31;
    const int w    = tid >> 5;                 // wave 0..15
    const int b0   = blockIdx.x * BT;
    const int hi   = (lane >= 16) ? 1 : 0;
    const int am   = lane & 15;

    // ---- one-time setup: cache ks<4 half of W_hh fragments in LDS ----
    {
        const uint4* src = (const uint4*)Wpk;
        uint4* dst = (uint4*)bc;
        for (int i = tid; i < BC_BYTES / 16; i += NTHREADS) {
            int fh     = i >> 6;          // cached frag id (64 uint4 each)
            int within = i & 63;
            int ntg    = fh >> 2;
            int ks     = fh & 3;
            dst[i] = src[(ntg * 8 + ks) * 64 + within];
        }
    }
    for (int i = tid; i < BT * HSTR; i += NTHREADS) h_lds[i] = (__bf16)0.0f;

    // cell state: c[r] for hidden j = w*16+am, batch M = r+8*hi
    float c[8];
    #pragma unroll
    for (int r = 0; r < 8; ++r) c[r] = 0.0f;
    const int j = w * 16 + am;

    if (tid < BT * INP) {   // preload x for t = 0
        int m = tid / INP, i = tid % INP;
        x_s[m * INP + i] = x[((size_t)(b0 + m) * T_STEPS + 0) * INP + i];
    }
    __syncthreads();

    unsigned long long wpu = (unsigned long long)Wpk;  // laundered each step
    int zs = 0;                                        // laundered each step

    for (int t = 0; t < T_STEPS; ++t) {
        // opaque redefinitions: block LICM from hoisting the (actually
        // loop-invariant) weight-fragment loads out of the t-loop
        asm volatile("" : "+s"(wpu), "+s"(zs));
        const __bf16* wp = (const __bf16*)wpu;

        // ---- build A_x fragment: x_t row (K=6) padded to K=32, 1.0 at K=6
        v16bf ax;
        {
            float x0 = x_s[am * INP + 0], x1 = x_s[am * INP + 1];
            float x2 = x_s[am * INP + 2], x3 = x_s[am * INP + 3];
            float x4 = x_s[am * INP + 4], x5 = x_s[am * INP + 5];
            const __bf16 zb = (__bf16)0.0f;
            ax[0] = hi ? zb : (__bf16)x0;
            ax[1] = hi ? zb : (__bf16)x1;
            ax[2] = hi ? zb : (__bf16)x2;
            ax[3] = hi ? zb : (__bf16)x3;
            ax[4] = hi ? zb : (__bf16)x4;
            ax[5] = hi ? zb : (__bf16)x5;
            ax[6] = hi ? zb : (__bf16)1.0f;   // bias row selector
            #pragma unroll
            for (int q = 7; q < 16; ++q) ax[q] = zb;
        }

        // ---- acc = xg + bias via one WMMA per tile (C = 0) ----
        v8f acc[4];                            // tile g
        const v8f zf = {0.f,0.f,0.f,0.f,0.f,0.f,0.f,0.f};
        #pragma unroll
        for (int g = 0; g < 4; ++g) {
            const int ntg = g * 16 + w;
            const v16bf bw = *(const v16bf*)(wp +
                  ((size_t)WHH_ELEMS + (size_t)ntg * 512 + (size_t)lane * 16));
            acc[g] = __builtin_amdgcn_wmma_f32_16x16x32_bf16(
                false, ax, false, bw, (short)0, zf, false, false);
        }

        // ---- acc += h @ W_hh^T : 8 k-steps of 32 ----
        #pragma unroll
        for (int ks = 0; ks < 8; ++ks) {
            union { v16bf v; unsigned u[8]; } a;
            #pragma unroll
            for (int p = 0; p < 8; ++p) {
                const int k_off = ((p < 4) ? (2 * p) : (16 + 2 * (p - 4))) + 8 * hi;
                const int idx   = am * HSTR + ks * 32 + k_off;   // even -> b32
                a.u[p] = *(const unsigned*)&h_lds[idx];
            }
            #pragma unroll
            for (int g = 0; g < 4; ++g) {
                const int ntg = g * 16 + w;
                v16bf bf;
                if (ks < 4) {
                    // LDS-cached half (offset laundered via zs)
                    bf = *(const v16bf*)((char*)bc +
                          (((ntg * 4 + ks) * 32 + lane) * 32 + zs));
                } else {
                    // L2-streamed half (pointer laundered via wpu)
                    bf = *(const v16bf*)(wp +
                          ((size_t)((ntg * 8 + ks) * 32 + lane) * 16));
                }
                acc[g] = __builtin_amdgcn_wmma_f32_16x16x32_bf16(
                    false, a.v, false, bf, (short)0, acc[g], false, false);
            }
        }
        __syncthreads();   // all waves done reading h_lds / x_s

        // prefetch x for t+1
        if (t + 1 < T_STEPS && tid < BT * INP) {
            int m = tid / INP, i = tid % INP;
            x_s[m * INP + i] = x[((size_t)(b0 + m) * T_STEPS + (t + 1)) * INP + i];
        }

        // ---- register-local elementwise: i,f,g,o -> c,h ----
        #pragma unroll
        for (int r = 0; r < 8; ++r) {
            const int M = r + 8 * hi;
            float ig = acc[0][r];
            float fg = acc[1][r];
            float gg = acc[2][r];
            float og = acc[3][r];
            ig = 1.0f / (1.0f + __expf(-ig));
            fg = 1.0f / (1.0f + __expf(-fg));
            gg = 2.0f / (1.0f + __expf(-2.0f * gg)) - 1.0f;   // tanh
            og = 1.0f / (1.0f + __expf(-og));
            float cv = fg * c[r] + ig * gg;
            c[r] = cv;
            float th = 2.0f / (1.0f + __expf(-2.0f * cv)) - 1.0f;  // tanh
            float hv = og * th;
            h_lds[M * HSTR + j] = (__bf16)hv;
        }
        __syncthreads();   // h_lds / x_s ready for next step
    }

    // ---- heads: 16 rows x 10 outputs per workgroup ----
    if (tid < BT * 10) {
        const int m  = tid / 10;
        const int hd = tid % 10;
        const float* Wp; float bb; int ob;
        if (hd < 2)      { Wp = Wg + hd * HID;       bb = bg[hd];     ob =        (b0 + m) * 2 + hd;       }
        else if (hd < 4) { Wp = Wh + (hd - 2) * HID; bb = bh[hd - 2]; ob = 256  + (b0 + m) * 2 + (hd - 2); }
        else if (hd < 7) { Wp = Wy + (hd - 4) * HID; bb = by[hd - 4]; ob = 512  + (b0 + m) * 3 + (hd - 4); }
        else             { Wp = Wl + (hd - 7) * HID; bb = bl[hd - 7]; ob = 896  + (b0 + m) * 3 + (hd - 7); }
        float s = bb;
        for (int k = 0; k < HID; ++k)
            s += (float)h_lds[m * HSTR + k] * Wp[k];
        out[ob] = s;
    }
}

extern "C" void kernel_launch(void* const* d_in, const int* in_sizes, int n_in,
                              void* d_out, int out_size, void* d_ws, size_t ws_size,
                              hipStream_t stream) {
    (void)in_sizes; (void)n_in; (void)out_size; (void)ws_size;
    const float* x    = (const float*)d_in[0];
    const float* W_ih = (const float*)d_in[1];
    const float* W_hh = (const float*)d_in[2];
    const float* b_ih = (const float*)d_in[3];
    const float* b_hh = (const float*)d_in[4];
    const float* Wg   = (const float*)d_in[5];
    const float* bg   = (const float*)d_in[6];
    const float* Wh   = (const float*)d_in[7];
    const float* bh   = (const float*)d_in[8];
    const float* Wy   = (const float*)d_in[9];
    const float* by   = (const float*)d_in[10];
    const float* Wl   = (const float*)d_in[11];
    const float* bl   = (const float*)d_in[12];

    __bf16* pk = (__bf16*)d_ws;   // 576 KB packed fragments

    static bool attr_set = false;
    if (!attr_set) {
        hipFuncSetAttribute((const void*)lstm_fused_kernel,
                            hipFuncAttributeMaxDynamicSharedMemorySize,
                            SMEM_BYTES);
        attr_set = true;
    }

    pack_weights_kernel<<<(PK_ELEMS + 255) / 256, 256, 0, stream>>>(
        W_hh, W_ih, b_ih, b_hh, pk);
    lstm_fused_kernel<<<128 / BT, NTHREADS, SMEM_BYTES, stream>>>(
        x, pk, Wg, bg, Wh, bh, Wy, by, Wl, bl, (float*)d_out);
}
